// GATEncoder_67765993997193
// MI455X (gfx1250) — matrix operand order repaired and checked by reading, not compiled, not run on hardware
//
#include <hip/hip_runtime.h>
#include <hip/hip_bf16.h>
#include <math.h>

#define NODES 50000
#define EDGES 800000
#define IN_CH 128
// layer1: heads=4, C=64 -> h1 [N, 256];  layer2: heads=1, C=64 -> h2 [N, 64]

typedef __attribute__((ext_vector_type(16))) __bf16 v16bf;
typedef __attribute__((ext_vector_type(8)))  __bf16 v8bf;
typedef __attribute__((ext_vector_type(8)))  float  v8f;

// ---------- helpers ----------
static __device__ __forceinline__ unsigned short f2bf(float f) {
  unsigned u = __float_as_uint(f);
  u += 0x7FFFu + ((u >> 16) & 1u);   // round-to-nearest-even
  return (unsigned short)(u >> 16);
}
// monotone float -> uint encoding so atomicMax(u32) == float max
static __device__ __forceinline__ unsigned enc_ord(float f) {
  unsigned u = __float_as_uint(f);
  return (u & 0x80000000u) ? ~u : (u | 0x80000000u);
}
static __device__ __forceinline__ float dec_ord(unsigned u) {
  return (u & 0x80000000u) ? __uint_as_float(u & 0x7FFFFFFFu) : __uint_as_float(~u);
}

// ---------- generic fills ----------
__global__ void fill_u32(unsigned* p, unsigned v, long long n) {
  long long i = (long long)blockIdx.x * blockDim.x + threadIdx.x;
  long long st = (long long)gridDim.x * blockDim.x;
  for (; i < n; i += st) p[i] = v;
}

// ---------- conversions ----------
__global__ void cvt_bf16(const float* __restrict__ s, unsigned short* __restrict__ d, int n) {
  int i = blockIdx.x * blockDim.x + threadIdx.x;
  if (i < n) d[i] = f2bf(s[i]);
}
// src [rows, cols] row-major -> dst [cols, rows] bf16 (B^T for WMMA)
__global__ void transpose_bf16(const float* __restrict__ s, unsigned short* __restrict__ d,
                               int rows, int cols) {
  int i = blockIdx.x * blockDim.x + threadIdx.x;
  if (i >= rows * cols) return;
  int r = i / cols, c = i % cols;
  d[(size_t)c * rows + r] = f2bf(s[i]);
}

// ---------- WMMA bf16 GEMM: C[M,Ntot] = A[M,K] * B[K,Ntot], BT = B^T row-major ----------
__global__ void wmma_gemm_bf16(const unsigned short* __restrict__ A,
                               const unsigned short* __restrict__ BT,
                               float* __restrict__ C,
                               int M, int K, int Ntot, int ntn) {
  const int gw   = (int)((blockIdx.x * blockDim.x + threadIdx.x) >> 5);
  const int lane = threadIdx.x & 31;
  const int total = (M >> 4) * ntn;
  if (gw >= total) return;                    // wave-uniform: EXEC stays all-ones for WMMA
  const int mt = gw / ntn, nt = gw % ntn;
  const int half = lane >> 4, l16 = lane & 15;
  const unsigned short* ap = A  + (size_t)(mt * 16 + l16) * K;
  const unsigned short* bp = BT + (size_t)(nt * 16 + l16) * K;
  v8f acc = {};
  for (int kb = 0; kb < K; kb += 32) {
    // A 16x32 bf16 frag: lanes 0-15 hold K {kb..kb+7, kb+16..kb+23}; lanes 16-31 the other blocks
    v8bf alo = *(const v8bf*)(ap + kb + half * 8);
    v8bf ahi = *(const v8bf*)(ap + kb + 16 + half * 8);
    v16bf a;
#pragma unroll
    for (int i = 0; i < 8; ++i) { a[i] = alo[i]; a[i + 8] = ahi[i]; }
    // B 32x16 bf16 frag: same K order as A (lanes 0-15 K {kb..kb+7, kb+16..kb+23}; lanes 16-31 the other blocks)
    v8bf blo = *(const v8bf*)(bp + kb + half * 8);
    v8bf bhi = *(const v8bf*)(bp + kb + 16 + half * 8);
    v16bf b;
#pragma unroll
    for (int i = 0; i < 8; ++i) { b[i] = blo[i]; b[i + 8] = bhi[i]; }
    acc = __builtin_amdgcn_wmma_f32_16x16x32_bf16(false, a, false, b, (short)0, acc, false, false);
  }
  float* cp = C + (size_t)(mt * 16 + half * 8) * Ntot + nt * 16 + l16;
#pragma unroll
  for (int r = 0; r < 8; ++r) cp[(size_t)r * Ntot] = acc[r];
}

// ---------- attention coefficients, layer 1 (4 heads x 64ch): wave per node ----------
__global__ void att_reduce1(const float* __restrict__ h1, const float* __restrict__ att_s,
                            const float* __restrict__ att_d,
                            float* __restrict__ a_s, float* __restrict__ a_d) {
  int w = (int)((blockIdx.x * blockDim.x + threadIdx.x) >> 5);
  int lane = threadIdx.x & 31;
  if (w >= NODES) return;
  int head = lane >> 3, l8 = lane & 7;          // 8 lanes per head
  const float* hp = h1 + (size_t)w * 256 + head * 64;
  float ss = 0.f, dd = 0.f;
#pragma unroll
  for (int i = 0; i < 8; ++i) {
    int c = l8 * 8 + i;
    float v = hp[c];
    ss += v * att_s[head * 64 + c];
    dd += v * att_d[head * 64 + c];
  }
  ss += __shfl_xor(ss, 1); dd += __shfl_xor(dd, 1);
  ss += __shfl_xor(ss, 2); dd += __shfl_xor(dd, 2);
  ss += __shfl_xor(ss, 4); dd += __shfl_xor(dd, 4);
  if (l8 == 0) { a_s[w * 4 + head] = ss; a_d[w * 4 + head] = dd; }
}

// ---------- attention coefficients, layer 2 (1 head x 64ch): wave per node ----------
__global__ void att_reduce2(const float* __restrict__ h2, const float* __restrict__ att_s,
                            const float* __restrict__ att_d,
                            float* __restrict__ a_s, float* __restrict__ a_d) {
  int w = (int)((blockIdx.x * blockDim.x + threadIdx.x) >> 5);
  int lane = threadIdx.x & 31;
  if (w >= NODES) return;
  float v0 = h2[(size_t)w * 64 + lane], v1 = h2[(size_t)w * 64 + lane + 32];
  float ss = v0 * att_s[lane] + v1 * att_s[lane + 32];
  float dd = v0 * att_d[lane] + v1 * att_d[lane + 32];
#pragma unroll
  for (int m = 16; m >= 1; m >>= 1) { ss += __shfl_xor(ss, m); dd += __shfl_xor(dd, m); }
  if (lane == 0) { a_s[w] = ss; a_d[w] = dd; }
}

// ---------- edge segment-max, layer 1: thread per (edge, head) ----------
__global__ void edge_max1(const int* __restrict__ ei0, const int* __restrict__ ei1,
                          const float* __restrict__ a_s, const float* __restrict__ a_d,
                          unsigned* __restrict__ m) {
  int i = blockIdx.x * blockDim.x + threadIdx.x;
  if (i >= (EDGES + NODES) * 4) return;
  int e = i >> 2, h = i & 3;
  int src = e < EDGES ? ei0[e] : e - EDGES;
  int dst = e < EDGES ? ei1[e] : e - EDGES;
  float sc = a_s[src * 4 + h] + a_d[dst * 4 + h];
  sc = sc > 0.f ? sc : 0.2f * sc;
  atomicMax(&m[dst * 4 + h], enc_ord(sc));
}

// ---------- fused edge exp + accumulate, layer 1: wave per edge ----------
__global__ void edge_acc1(const int* __restrict__ ei0, const int* __restrict__ ei1,
                          const float* __restrict__ a_s, const float* __restrict__ a_d,
                          const unsigned* __restrict__ m, const float* __restrict__ h1,
                          float* __restrict__ s, float* __restrict__ acc) {
  int w = (int)((blockIdx.x * blockDim.x + threadIdx.x) >> 5);
  int lane = threadIdx.x & 31;
  if (w >= EDGES + NODES) return;
  int src = w < EDGES ? ei0[w] : w - EDGES;
  int dst = w < EDGES ? ei1[w] : w - EDGES;
  const float* hs = h1 + (size_t)src * 256;
  float* ap = acc + (size_t)dst * 256;
#pragma unroll
  for (int h = 0; h < 4; ++h) {
    float sc = a_s[src * 4 + h] + a_d[dst * 4 + h];
    sc = sc > 0.f ? sc : 0.2f * sc;
    float mx = dec_ord(m[dst * 4 + h]);
    if (!isfinite(mx)) mx = 0.f;
    float p = __expf(sc - mx);
    if (lane == h) atomicAdd(&s[dst * 4 + h], p);
    atomicAdd(&ap[h * 64 + lane],      p * hs[h * 64 + lane]);
    atomicAdd(&ap[h * 64 + lane + 32], p * hs[h * 64 + lane + 32]);
  }
}

// ---------- finalize layer 1: mean over heads + bias + ELU -> bf16 ----------
__global__ void finalize1(const float* __restrict__ acc, const float* __restrict__ s,
                          const float* __restrict__ b1, unsigned short* __restrict__ hm) {
  int i = blockIdx.x * blockDim.x + threadIdx.x;
  if (i >= NODES * 64) return;
  int n = i >> 6, c = i & 63;
  float v = 0.f;
#pragma unroll
  for (int h = 0; h < 4; ++h)
    v += acc[(size_t)n * 256 + h * 64 + c] / (s[n * 4 + h] + 1e-16f);
  v = 0.25f * v + b1[c];
  v = v > 0.f ? v : expm1f(v);
  hm[i] = f2bf(v);
}

// ---------- edge segment-max, layer 2: thread per edge ----------
__global__ void edge_max2(const int* __restrict__ ei0, const int* __restrict__ ei1,
                          const float* __restrict__ a_s, const float* __restrict__ a_d,
                          unsigned* __restrict__ m) {
  int e = blockIdx.x * blockDim.x + threadIdx.x;
  if (e >= EDGES + NODES) return;
  int src = e < EDGES ? ei0[e] : e - EDGES;
  int dst = e < EDGES ? ei1[e] : e - EDGES;
  float sc = a_s[src] + a_d[dst];
  sc = sc > 0.f ? sc : 0.2f * sc;
  atomicMax(&m[dst], enc_ord(sc));
}

// ---------- fused edge exp + accumulate, layer 2: wave per edge ----------
__global__ void edge_acc2(const int* __restrict__ ei0, const int* __restrict__ ei1,
                          const float* __restrict__ a_s, const float* __restrict__ a_d,
                          const unsigned* __restrict__ m, const float* __restrict__ h2,
                          float* __restrict__ s, float* __restrict__ acc) {
  int w = (int)((blockIdx.x * blockDim.x + threadIdx.x) >> 5);
  int lane = threadIdx.x & 31;
  if (w >= EDGES + NODES) return;
  int src = w < EDGES ? ei0[w] : w - EDGES;
  int dst = w < EDGES ? ei1[w] : w - EDGES;
  float sc = a_s[src] + a_d[dst];
  sc = sc > 0.f ? sc : 0.2f * sc;
  float mx = dec_ord(m[dst]);
  if (!isfinite(mx)) mx = 0.f;
  float p = __expf(sc - mx);
  if (lane == 0) atomicAdd(&s[dst], p);
  atomicAdd(&acc[(size_t)dst * 64 + lane],      p * h2[(size_t)src * 64 + lane]);
  atomicAdd(&acc[(size_t)dst * 64 + lane + 32], p * h2[(size_t)src * 64 + lane + 32]);
}

// ---------- finalize layer 2: bias + split into (mu, logstd) ----------
__global__ void finalize2(const float* __restrict__ acc, const float* __restrict__ s,
                          const float* __restrict__ b2, float* __restrict__ out) {
  int i = blockIdx.x * blockDim.x + threadIdx.x;
  if (i >= NODES * 64) return;
  int n = i >> 6, c = i & 63;
  float v = acc[i] / (s[n] + 1e-16f) + b2[c];
  if (c < 32) out[(size_t)n * 32 + c] = v;                              // mu
  else        out[(size_t)NODES * 32 + (size_t)n * 32 + (c - 32)] = v;  // logstd
}

extern "C" void kernel_launch(void* const* d_in, const int* in_sizes, int n_in,
                              void* d_out, int out_size, void* d_ws, size_t ws_size,
                              hipStream_t stream) {
  (void)in_sizes; (void)n_in; (void)out_size; (void)ws_size;
  const float* x   = (const float*)d_in[0];
  const int*   ei  = (const int*)d_in[1];
  const int *ei0 = ei, *ei1 = ei + EDGES;
  const float* W1  = (const float*)d_in[2];
  const float* as1 = (const float*)d_in[3];
  const float* ad1 = (const float*)d_in[4];
  const float* b1  = (const float*)d_in[5];
  const float* W2  = (const float*)d_in[6];
  const float* as2 = (const float*)d_in[7];
  const float* ad2 = (const float*)d_in[8];
  const float* b2  = (const float*)d_in[9];
  float* out = (float*)d_out;

  char* ws = (char*)d_ws;
  size_t off = 0;
  auto alloc = [&](size_t bytes) -> void* {
    void* p = ws + off; off = (off + bytes + 255) & ~(size_t)255; return p;
  };
  unsigned short* x_bf  = (unsigned short*)alloc((size_t)NODES * IN_CH * 2);
  unsigned short* W1T   = (unsigned short*)alloc((size_t)256 * IN_CH * 2);
  float*          h1    = (float*)alloc((size_t)NODES * 256 * 4);
  float*          aS1   = (float*)alloc((size_t)NODES * 4 * 4);
  float*          aD1   = (float*)alloc((size_t)NODES * 4 * 4);
  unsigned*       m1    = (unsigned*)alloc((size_t)NODES * 4 * 4);
  float*          s1    = (float*)alloc((size_t)NODES * 4 * 4);
  float*          acc1  = (float*)alloc((size_t)NODES * 256 * 4);
  unsigned short* hm_bf = (unsigned short*)alloc((size_t)NODES * 64 * 2);
  unsigned short* W2T   = (unsigned short*)alloc((size_t)64 * 64 * 2);
  float*          h2    = (float*)alloc((size_t)NODES * 64 * 4);
  float*          aS2   = (float*)alloc((size_t)NODES * 4);
  float*          aD2   = (float*)alloc((size_t)NODES * 4);
  unsigned*       m2    = (unsigned*)alloc((size_t)NODES * 4);
  float*          s2    = (float*)alloc((size_t)NODES * 4);
  float*          acc2  = (float*)alloc((size_t)NODES * 64 * 4);

  const int B = 256;
  // init: all init values (0.0f, and ordered-encoding minimum) are bit-pattern 0
  fill_u32<<<4096, B, 0, stream>>>(m1, 0u, (long long)NODES * 4);
  fill_u32<<<4096, B, 0, stream>>>((unsigned*)s1, 0u, (long long)NODES * 4);
  fill_u32<<<4096, B, 0, stream>>>((unsigned*)acc1, 0u, (long long)NODES * 256);
  fill_u32<<<4096, B, 0, stream>>>(m2, 0u, (long long)NODES);
  fill_u32<<<4096, B, 0, stream>>>((unsigned*)s2, 0u, (long long)NODES);
  fill_u32<<<4096, B, 0, stream>>>((unsigned*)acc2, 0u, (long long)NODES * 64);

  // ---- layer 1 GEMM: h1 = x @ W1 (bf16 WMMA, fp32 accumulate) ----
  cvt_bf16<<<(NODES * IN_CH + B - 1) / B, B, 0, stream>>>(x, x_bf, NODES * IN_CH);
  transpose_bf16<<<(IN_CH * 256 + B - 1) / B, B, 0, stream>>>(W1, W1T, IN_CH, 256);
  {
    int waves = (NODES / 16) * 16;                 // 3125 * 16 tiles
    wmma_gemm_bf16<<<(waves * 32 + B - 1) / B, B, 0, stream>>>(x_bf, W1T, h1,
                                                               NODES, IN_CH, 256, 16);
  }
  att_reduce1<<<(NODES * 32 + B - 1) / B, B, 0, stream>>>(h1, as1, ad1, aS1, aD1);
  edge_max1<<<((EDGES + NODES) * 4 + B - 1) / B, B, 0, stream>>>(ei0, ei1, aS1, aD1, m1);
  edge_acc1<<<((EDGES + NODES) * 32 + B - 1) / B, B, 0, stream>>>(ei0, ei1, aS1, aD1, m1,
                                                                  h1, s1, acc1);
  finalize1<<<(NODES * 64 + B - 1) / B, B, 0, stream>>>(acc1, s1, b1, hm_bf);

  // ---- layer 2 GEMM: h2 = h_mid @ W2 ----
  transpose_bf16<<<(64 * 64 + B - 1) / B, B, 0, stream>>>(W2, W2T, 64, 64);
  {
    int waves = (NODES / 16) * 4;                  // 3125 * 4 tiles
    wmma_gemm_bf16<<<(waves * 32 + B - 1) / B, B, 0, stream>>>(hm_bf, W2T, h2,
                                                               NODES, 64, 64, 4);
  }
  att_reduce2<<<(NODES * 32 + B - 1) / B, B, 0, stream>>>(h2, as2, ad2, aS2, aD2);
  edge_max2<<<((EDGES + NODES) + B - 1) / B, B, 0, stream>>>(ei0, ei1, aS2, aD2, m2);
  edge_acc2<<<((EDGES + NODES) * 32 + B - 1) / B, B, 0, stream>>>(ei0, ei1, aS2, aD2, m2,
                                                                  h2, s2, acc2);
  finalize2<<<(NODES * 64 + B - 1) / B, B, 0, stream>>>(acc2, s2, b2, out);
}
